// GATLayer_20873541058924
// MI455X (gfx1250) — matrix operand order repaired and checked by reading, not compiled, not run on hardware
//
#include <hip/hip_runtime.h>
#include <hip/hip_bf16.h>

#define NN   8192
#define INF_ 512
#define OF   256

typedef __attribute__((ext_vector_type(16))) __bf16 v16bf;
typedef __attribute__((ext_vector_type(8)))  float  v8f;
typedef __attribute__((ext_vector_type(4)))  unsigned int u32x4;
typedef __attribute__((ext_vector_type(8)))  int          i32x8;
typedef __attribute__((ext_vector_type(4)))  int          i32x4;

struct U4x2 { uint4 lo; uint4 hi; };   // 32 bytes == v16bf

__device__ __forceinline__ unsigned short bfbits(float f) {
    return __builtin_bit_cast(unsigned short, (__bf16)f);
}

// Low 32 bits of a flat shared pointer == wave-relative LDS byte offset
// (LDS aperture base has zero low 32 bits per CDNA5 aperture rules).
__device__ __forceinline__ unsigned lds_off(const void* p) {
    return (unsigned)(uintptr_t)p;
}

// CDNA5 async DMA: global -> LDS, 16B per lane, tracked by ASYNCcnt.
__device__ __forceinline__ void async_g2l_b128(unsigned lds, const void* g) {
    asm volatile("global_load_async_to_lds_b128 %0, %1, off"
                 :: "v"(lds), "v"(g) : "memory");
}
__device__ __forceinline__ void wait_async0() {
    asm volatile("s_wait_asynccnt 0x0" ::: "memory");
}

// A-fragment (16x32 bf16): contiguous runs K=8h..8h+7 and K=16+8h..23+8h.
__device__ __forceinline__ v16bf load_fragA(const __bf16* rowp, int hl) {
    U4x2 u;
    u.lo = *(const uint4*)(rowp + 8 * hl);
    u.hi = *(const uint4*)(rowp + 16 + 8 * hl);
    return __builtin_bit_cast(v16bf, u);
}
// B-fragment (32x16 bf16, LDS stored n-major): runs K=8h..8h+7 and K=16+8h..23+8h.
__device__ __forceinline__ v16bf load_fragB(const __bf16* colp, int hl) {
    U4x2 u;
    u.lo = *(const uint4*)(colp + 8 * hl);
    u.hi = *(const uint4*)(colp + 16 + 8 * hl);
    return __builtin_bit_cast(v16bf, u);
}

// ---------------------------------------------------------------------------
// Kernel 0: Wt[n][k] = bf16(W[k][n])  (256 KB, one-shot, L2-resident after)
// ---------------------------------------------------------------------------
__global__ __launch_bounds__(256) void k_wt(
    const float* __restrict__ W, __bf16* __restrict__ Wt)
{
    const int id = blockIdx.x * 256 + threadIdx.x;     // 0..32767
    const int k = id >> 6, n4 = (id & 63) * 4;
    const float4 v = *(const float4*)(W + (size_t)k * OF + n4);
    Wt[(size_t)(n4 + 0) * INF_ + k] = (__bf16)v.x;
    Wt[(size_t)(n4 + 1) * INF_ + k] = (__bf16)v.y;
    Wt[(size_t)(n4 + 2) * INF_ + k] = (__bf16)v.z;
    Wt[(size_t)(n4 + 3) * INF_ + k] = (__bf16)v.w;
}

// ---------------------------------------------------------------------------
// Kernel 1: h = x @ W (bf16 WMMA, f32 accum). Writes h (f32, row-major) and
// hbT (bf16, TRANSPOSED [OF][NN]) for the fused attention GEMM's TDM path.
// B tile staged from Wt via async global->LDS DMA (pure contiguous copy).
// ---------------------------------------------------------------------------
__global__ __launch_bounds__(256) void k_gemm_xw(
    const float* __restrict__ x, const __bf16* __restrict__ Wt,
    float* __restrict__ h, __bf16* __restrict__ hbT)
{
    __shared__ __bf16 As [32][40];    // 32(M) x 32(K), row stride 80B
    __shared__ __bf16 BsT[256][40];   // 256(N) x 32(K) n-major, stride 80B

    const int t    = threadIdx.x;
    const int wave = t >> 5, lane = t & 31;
    const int mt   = wave >> 2;       // 16-row half of the 32-row tile
    const int ng   = wave & 3;        // 64-col group
    const int hl   = lane >> 4, ln = lane & 15;
    const int i0   = blockIdx.x * 32;

    v8f cacc[4] = {};

    for (int k0 = 0; k0 < INF_; k0 += 32) {
        // Async-stage B: BsT[n][0..31] <- Wt[n][k0..k0+31] (1024 x 16B)
        #pragma unroll
        for (int q = 0; q < 4; ++q) {
            const int id = t + 256 * q;
            const int f  = id >> 2;
            const int cb = (id & 3) * 16;
            async_g2l_b128(lds_off(&BsT[f][0]) + cb,
                           (const char*)(Wt + (size_t)f * INF_ + k0) + cb);
        }
        // Stage A: x tile f32 -> bf16, packed 8B LDS stores (overlaps DMA)
        {
            const int r = t >> 3, c = (t & 7) * 4;
            const float4 v = *(const float4*)(x + (size_t)(i0 + r) * INF_ + k0 + c);
            uint2 p;
            p.x = (unsigned)bfbits(v.x) | ((unsigned)bfbits(v.y) << 16);
            p.y = (unsigned)bfbits(v.z) | ((unsigned)bfbits(v.w) << 16);
            *(uint2*)&As[r][c] = p;
        }
        wait_async0();
        __syncthreads();

        const v16bf a = load_fragA(&As[mt * 16 + ln][0], hl);
        #pragma unroll
        for (int nt = 0; nt < 4; ++nt) {
            const v16bf b = load_fragB(&BsT[ng * 64 + nt * 16 + ln][0], hl);
            cacc[nt] = __builtin_amdgcn_wmma_f32_16x16x32_bf16(
                false, a, false, b, (short)0, cacc[nt], false, false);
        }
        __syncthreads();
    }

    // Epilogue: h (f32 row-major) + hbT (bf16 f-major, 16B packed stores).
    #pragma unroll
    for (int nt = 0; nt < 4; ++nt) {
        const int col     = ng * 64 + nt * 16 + ln;
        const int rowbase = i0 + mt * 16 + 8 * hl;
        uint4 pk;
        pk.x = (unsigned)bfbits(cacc[nt][0]) | ((unsigned)bfbits(cacc[nt][1]) << 16);
        pk.y = (unsigned)bfbits(cacc[nt][2]) | ((unsigned)bfbits(cacc[nt][3]) << 16);
        pk.z = (unsigned)bfbits(cacc[nt][4]) | ((unsigned)bfbits(cacc[nt][5]) << 16);
        pk.w = (unsigned)bfbits(cacc[nt][6]) | ((unsigned)bfbits(cacc[nt][7]) << 16);
        *(uint4*)(hbT + (size_t)col * NN + rowbase) = pk;
        #pragma unroll
        for (int r = 0; r < 8; ++r)
            h[(size_t)(rowbase + r) * OF + col] = cacc[nt][r];
    }
}

// ---------------------------------------------------------------------------
// Kernel 2: s_src = h @ a[:256], s_dst = h @ a[256:]. One row per wave32.
// ---------------------------------------------------------------------------
__global__ __launch_bounds__(256) void k_svec(
    const float* __restrict__ h, const float* __restrict__ a,
    float* __restrict__ s_src, float* __restrict__ s_dst)
{
    const int t = threadIdx.x, wave = t >> 5, lane = t & 31;
    const int i = blockIdx.x * 8 + wave;
    float d1 = 0.f, d2 = 0.f;
    #pragma unroll
    for (int k = lane; k < OF; k += 32) {
        const float hv = h[(size_t)i * OF + k];
        d1 += hv * a[k];
        d2 += hv * a[OF + k];
    }
    #pragma unroll
    for (int off = 16; off > 0; off >>= 1) {
        d1 += __shfl_xor(d1, off, 32);
        d2 += __shfl_xor(d2, off, 32);
    }
    if (lane == 0) { s_src[i] = d1; s_dst[i] = d2; }
}

// ---------------------------------------------------------------------------
// Kernel 3: b[i] = max_j lrelu(s_src[i]+s_dst[j]) + log(sum_j exp(..-max)).
// ---------------------------------------------------------------------------
__global__ __launch_bounds__(256) void k_stats(
    const float* __restrict__ s_src, const float* __restrict__ s_dst,
    float* __restrict__ brow)
{
    __shared__ float sd[NN];
    __shared__ float pm[64][4];
    __shared__ float pz[64][4];
    const int t = threadIdx.x;
    #pragma unroll
    for (int q = 0; q < NN / 256; ++q) sd[t + 256 * q] = s_dst[t + 256 * q];
    __syncthreads();

    const int r = t >> 2, p = t & 3;
    const int i = blockIdx.x * 64 + r;
    const float ss = s_src[i];
    const int j0 = p * (NN / 4);

    float m = -3.0e38f;
    for (int j = 0; j < NN / 4; ++j) {
        float v = ss + sd[j0 + j];
        v = v > 0.f ? v : 0.2f * v;
        m = fmaxf(m, v);
    }
    float z = 0.f;
    for (int j = 0; j < NN / 4; ++j) {
        float v = ss + sd[j0 + j];
        v = v > 0.f ? v : 0.2f * v;
        z += __expf(v - m);
    }
    pm[r][p] = m; pz[r][p] = z;
    __syncthreads();
    if (p == 0) {
        const float M = fmaxf(fmaxf(pm[r][0], pm[r][1]), fmaxf(pm[r][2], pm[r][3]));
        float Z = 0.f;
        #pragma unroll
        for (int q = 0; q < 4; ++q) Z += pz[r][q] * __expf(pm[r][q] - M);
        brow[i] = M + __logf(Z);
    }
}

// ---------------------------------------------------------------------------
// Kernel 4 (fused): attention tile -> d_out, h_prime += att @ h (bf16 WMMA).
// B operand staged by the Tensor Data Mover: one tensor_load_to_lds per tile
// (2D tile 256 rows x 64B, HW-padded +16B/row to match BsT's 80B stride),
// issued by wave 0, tracked by TENSORcnt, overlapped with exp/lrelu VALU.
// ---------------------------------------------------------------------------
__global__ __launch_bounds__(256) void k_att(
    const float* __restrict__ adj, const float* __restrict__ s_src,
    const float* __restrict__ s_dst, const float* __restrict__ brow,
    const __bf16* __restrict__ hbT,
    float* __restrict__ out_hp, float* __restrict__ out_att)
{
    __shared__ __bf16 As [32][40];    // 32(M) x 32(K) attention tile
    __shared__ __bf16 BsT[256][40];   // 256(N=f) x 32(K=j) h tile, n-major

    const int t    = threadIdx.x;
    const int wave = t >> 5, lane = t & 31;
    const int mt   = wave >> 2, ng = wave & 3;
    const int hl   = lane >> 4, ln = lane & 15;
    const int i0   = blockIdx.x * 32;

    const int sr = t >> 3;          // staging row within tile (0..31), fixed
    const int sc = (t & 7) * 4;     // staging col (0..28 step 4)
    const float ss = s_src[i0 + sr];
    const float bb = brow[i0 + sr];

    // TDM descriptor pieces that don't depend on j0.
    // Group1: data_size=2B(code1)@16 | pad_enable@20 | pad_interval=64B(3)@22
    //         | pad_amount=16B(3)@25 ; tensor_dim0=NN ; tensor_dim1=OF ;
    //         tile_dim0=32 ; tile_dim1=256 ; tensor_dim0_stride=NN.
    i32x8 g1;
    g1[0] = (int)((1u << 16) | (1u << 20) | (3u << 22) | (3u << 25));
    g1[1] = (int)(((unsigned)NN & 0xffffu) << 16);      // dim0 low16 @ [31:16]
    g1[2] = (int)((NN >> 16) | (((unsigned)OF & 0xffffu) << 16));
    g1[3] = (int)((OF >> 16) | (32u << 16));            // tile_dim0=32
    g1[4] = 256;                                        // tile_dim1=256
    g1[5] = NN;                                         // dim0_stride low32
    g1[6] = 0;
    g1[7] = 0;
    const i32x4 gz4 = {0, 0, 0, 0};
    const i32x8 gz8 = {0, 0, 0, 0, 0, 0, 0, 0};
    const unsigned ldsB = lds_off(&BsT[0][0]);
    const unsigned long long gbase = (unsigned long long)(uintptr_t)hbT;

    v8f cacc[4] = {};

    for (int j0 = 0; j0 < NN; j0 += 32) {
        // Wave 0 kicks the TDM for the whole 16KB B tile.
        if (wave == 0) {
            const unsigned long long ga = gbase + (unsigned long long)j0 * 2u;
            u32x4 g0;
            g0[0] = 1u;                                  // count=1 descriptor
            g0[1] = ldsB;                                // lds_addr
            g0[2] = (unsigned)(ga & 0xffffffffu);        // global_addr lo
            g0[3] = (unsigned)(ga >> 32) | (2u << 30);   // global_addr hi|type=2
            __builtin_amdgcn_tensor_load_to_lds(g0, g1, gz4, gz4, gz8, 0);
        }
        if (j0 + 32 < NN)
            __builtin_prefetch(adj + (size_t)(i0 + sr) * NN + j0 + 32 + sc, 0, 0);

        // A tile: attention values (overlaps the TDM transfer)
        {
            const float4 av = *(const float4*)(adj + (size_t)(i0 + sr) * NN + j0 + sc);
            const float4 sv = *(const float4*)(s_dst + j0 + sc);
            float4 att; float v;
            v = ss + sv.x; v = v > 0.f ? v : 0.2f * v; att.x = __expf(v - bb) * av.x;
            v = ss + sv.y; v = v > 0.f ? v : 0.2f * v; att.y = __expf(v - bb) * av.y;
            v = ss + sv.z; v = v > 0.f ? v : 0.2f * v; att.z = __expf(v - bb) * av.z;
            v = ss + sv.w; v = v > 0.f ? v : 0.2f * v; att.w = __expf(v - bb) * av.w;
            *(float4*)(out_att + (size_t)(i0 + sr) * NN + j0 + sc) = att;
            uint2 p;
            p.x = (unsigned)bfbits(att.x) | ((unsigned)bfbits(att.y) << 16);
            p.y = (unsigned)bfbits(att.z) | ((unsigned)bfbits(att.w) << 16);
            *(uint2*)&As[sr][sc] = p;
        }

        if (wave == 0) __builtin_amdgcn_s_wait_tensorcnt((short)0);
        __syncthreads();

        const v16bf a = load_fragA(&As[mt * 16 + ln][0], hl);
        #pragma unroll
        for (int nt = 0; nt < 4; ++nt) {
            const v16bf b = load_fragB(&BsT[ng * 64 + nt * 16 + ln][0], hl);
            cacc[nt] = __builtin_amdgcn_wmma_f32_16x16x32_bf16(
                false, a, false, b, (short)0, cacc[nt], false, false);
        }
        __syncthreads();
    }

    #pragma unroll
    for (int nt = 0; nt < 4; ++nt) {
        const int col = ng * 64 + nt * 16 + ln;
        #pragma unroll
        for (int r = 0; r < 8; ++r) {
            const int row = i0 + mt * 16 + r + 8 * hl;
            out_hp[(size_t)row * OF + col] = cacc[nt][r];
        }
    }
}

// ---------------------------------------------------------------------------
extern "C" void kernel_launch(void* const* d_in, const int* in_sizes, int n_in,
                              void* d_out, int out_size, void* d_ws, size_t ws_size,
                              hipStream_t stream) {
    (void)in_sizes; (void)n_in; (void)out_size; (void)ws_size;

    const float* x   = (const float*)d_in[0];   // [8192, 512]
    const float* adj = (const float*)d_in[1];   // [8192, 8192]
    const float* W   = (const float*)d_in[2];   // [512, 256]
    const float* a   = (const float*)d_in[3];   // [512, 1]

    float* out     = (float*)d_out;
    float* out_hp  = out;                          // [8192, 256]
    float* out_att = out + (size_t)NN * OF;        // [8192, 8192]

    // Workspace layout (~12.9 MB)
    float*  h     = (float*)d_ws;                          // 8 MB
    __bf16* hbT   = (__bf16*)(h + (size_t)NN * OF);        // 4 MB, [OF][NN]
    float*  s_src = (float*)(hbT + (size_t)NN * OF);       // 32 KB
    float*  s_dst = s_src + NN;                            // 32 KB
    float*  brow  = s_dst + NN;                            // 32 KB
    __bf16* Wt    = (__bf16*)(brow + NN);                  // 256 KB, [OF][INF_]

    k_wt     <<<128,     256, 0, stream>>>(W, Wt);
    k_gemm_xw<<<NN / 32, 256, 0, stream>>>(x, Wt, h, hbT);
    k_svec   <<<NN / 8,  256, 0, stream>>>(h, a, s_src, s_dst);
    k_stats  <<<NN / 64, 256, 0, stream>>>(s_src, s_dst, brow);
    k_att    <<<NN / 32, 256, 0, stream>>>(adj, s_src, s_dst, brow, hbT,
                                           out_hp, out_att);
}
